// NeuralCellularAutomata_17437567222038
// MI455X (gfx1250) — matrix operand order repaired and checked by reading, not compiled, not run on hardware
//
#include <hip/hip_runtime.h>
#include <math.h>

// ---------------------------------------------------------------------------
// NCA step fused kernel for MI455X (gfx1250, wave32, WMMA bf16 16x16x32)
//   x:[16,128,128,64] f32 -> 4x depthwise dilated 3x3 (d=1,2,4,8)
//   concat -> [.,320] -> Linear(320,256)+GELU -> Linear(256,64) -> +x
// M-blocked: 32 pixels (2 WMMA M-tiles) per workgroup so each W1t B-operand
// fetch feeds 2 WMMAs -> halves L2 weight traffic per pixel.
// ---------------------------------------------------------------------------

typedef __attribute__((ext_vector_type(16))) __bf16 v16bf;
typedef __attribute__((ext_vector_type(8)))  float  v8f;

#define IMG_H   128
#define IMG_W   128
#define CH      64
#define KIN     320          // 5*CH concatenated features
#define HID     256
#define PST     336          // padded LDS row stride (halves) for P tile
#define HST     272          // padded LDS row stride (halves) for H tile
#define TILE_M  32           // pixels per workgroup (2 WMMA M tiles)

__device__ __forceinline__ unsigned short f2bf(float f) {
    union { float f; unsigned u; } v; v.f = f;
    unsigned u = v.u;
    u += 0x7FFFu + ((u >> 16) & 1u);   // round-to-nearest-even
    return (unsigned short)(u >> 16);
}

__device__ __forceinline__ float gelu_exact(float x) {
    return 0.5f * x * (1.0f + erff(x * 0.70710678118654752f));
}

// Build a bf16 A/B operand vector from two 16-byte chunks.
__device__ __forceinline__ v16bf pack_v16bf(uint4 lo, uint4 hi) {
    union { uint4 q[2]; v16bf v; } u;
    u.q[0] = lo; u.q[1] = hi;
    return u.v;
}

#define WMMA_BF16(A, B, C) \
    __builtin_amdgcn_wmma_f32_16x16x32_bf16(false, (A), false, (B), (short)0, (C), false, false)

// ----------------------------- weight prep --------------------------------
// W1 [320][256] f32 -> W1t bf16 [256][320] ; W2 [256][64] f32 -> W2t [64][256]
__global__ void nca_prep_weights(const float* __restrict__ W1,
                                 const float* __restrict__ W2,
                                 unsigned short* __restrict__ W1t,
                                 unsigned short* __restrict__ W2t) {
    int i = blockIdx.x * 256 + threadIdx.x;
    if (i < KIN * HID) {                      // 81920
        int n = i / KIN, k = i - n * KIN;
        W1t[i] = f2bf(W1[k * HID + n]);
    } else {
        int o = i - KIN * HID;                // 16384 entries
        if (o < HID * CH) {
            int n = o / HID, k = o - n * HID;
            W2t[o] = f2bf(W2[k * CH + n]);
        }
    }
}

// ------------------------------ main kernel -------------------------------
__global__ void __launch_bounds__(256)
nca_step_kernel(const float* __restrict__ x,
                const float* __restrict__ conv_w,   // [4][3][3][1][64]
                const float* __restrict__ b1,       // [256]
                const float* __restrict__ b2,       // [64]
                const unsigned short* __restrict__ W1t,  // bf16 [256][320]
                const unsigned short* __restrict__ W2t,  // bf16 [64][256]
                float* __restrict__ out) {
    __shared__ __align__(16) unsigned short Psh[TILE_M * PST]; // 32x320 features (bf16)
    __shared__ __align__(16) unsigned short Hsh[TILE_M * HST]; // 32x256 hidden  (bf16)

    const int tid  = threadIdx.x;
    const int tile = blockIdx.x;                 // 262144/32 = 8192 tiles
    // tile -> (b, h, w0); 4 tiles per image row (32 consecutive pixels each)
    const int bimg = tile >> 9;                  // 512 tiles per image
    const int rem  = tile & 511;
    const int hrow = rem >> 2;
    const int w0   = (rem & 3) << 5;
    const float* xb = x + (size_t)bimg * IMG_H * IMG_W * CH;

    // ---------------- phase 1: perception (all 8 waves) ----------------
    {
        const int c = tid & 63;                  // channel
        const int q = tid >> 6;                  // pixel group 0..3 (8 px each)
        float wr[4][9];
        #pragma unroll
        for (int i = 0; i < 4; ++i)
            #pragma unroll
            for (int t = 0; t < 9; ++t)
                wr[i][t] = conv_w[(i * 9 + t) * CH + c];

        #pragma unroll
        for (int pi = 0; pi < 8; ++pi) {
            const int p    = q * 8 + pi;
            const int wcol = w0 + p;
            // identity feature (center pixel), reused as the center tap below
            const float xc = xb[(hrow * IMG_W + wcol) * CH + c];
            Psh[p * PST + c] = f2bf(xc);
            // 4 dilated depthwise 3x3; zero padding at borders.
            // All gathers are L2 hits: full x (64MB) lives in the 192MB L2.
            #pragma unroll
            for (int i = 0; i < 4; ++i) {
                const int dil = 1 << i;
                float acc = xc * wr[i][4];       // center tap
                #pragma unroll
                for (int ky = 0; ky < 3; ++ky) {
                    const int hh = hrow + (ky - 1) * dil;
                    if (hh < 0 || hh >= IMG_H) continue;
                    #pragma unroll
                    for (int kx = 0; kx < 3; ++kx) {
                        if (ky == 1 && kx == 1) continue;
                        const int ww = wcol + (kx - 1) * dil;
                        if (ww < 0 || ww >= IMG_W) continue;
                        acc += xb[(hh * IMG_W + ww) * CH + c] * wr[i][ky * 3 + kx];
                    }
                }
                Psh[p * PST + CH + i * CH + c] = f2bf(acc);
            }
        }
    }
    __syncthreads();

    // common WMMA lane decomposition (wave32)
    const int lane = tid & 31;
    const int wv   = tid >> 5;       // wave 0..7
    const int hi   = lane >> 4;      // half-wave select
    const int col  = lane & 15;      // N column within 16x16 tile / A row

    // ------- phase 2: GEMM1 [32x320]x[320x256], 2 M-tiles per wave -------
    // wave wv owns hidden columns [wv*32, wv*32+32); each B fetch feeds 2 WMMAs
    {
        const int nb0 = wv * 32;
        const float bv0 = b1[nb0 + col];
        const float bv1 = b1[nb0 + 16 + col];
        v8f a00, a01, a10, a11;                  // [mtile][nblock]
        #pragma unroll
        for (int i = 0; i < 8; ++i) {
            a00[i] = bv0; a01[i] = bv1;
            a10[i] = bv0; a11[i] = bv1;
        }

        #pragma unroll
        for (int ks = 0; ks < KIN / 32; ++ks) {          // 10 K-steps
            const int kO = ks * 32 + hi * 8;             // A chunk base (halves)
            const unsigned short* ap0 = &Psh[col * PST + kO];          // rows 0..15
            const unsigned short* ap1 = &Psh[(16 + col) * PST + kO];   // rows 16..31
            v16bf A0 = pack_v16bf(*(const uint4*)ap0, *(const uint4*)(ap0 + 16));
            v16bf A1 = pack_v16bf(*(const uint4*)ap1, *(const uint4*)(ap1 + 16));
            const int kb = ks * 32 + hi * 8;             // B: two 8-K chunks (table order)
            const unsigned short* bp0 = W1t + (size_t)(nb0 + col) * KIN + kb;
            const unsigned short* bp1 = W1t + (size_t)(nb0 + 16 + col) * KIN + kb;
            v16bf B0 = pack_v16bf(*(const uint4*)bp0, *(const uint4*)(bp0 + 16));
            v16bf B1 = pack_v16bf(*(const uint4*)bp1, *(const uint4*)(bp1 + 16));
            a00 = WMMA_BF16(A0, B0, a00);
            a10 = WMMA_BF16(A1, B0, a10);
            a01 = WMMA_BF16(A0, B1, a01);
            a11 = WMMA_BF16(A1, B1, a11);
        }

        // exact GELU + bf16 convert, stage hidden tile to LDS
        #pragma unroll
        for (int r = 0; r < 8; ++r) {
            const int m = r + hi * 8;                    // row within M-tile
            Hsh[m * HST + nb0 + col]             = f2bf(gelu_exact(a00[r]));
            Hsh[m * HST + nb0 + 16 + col]        = f2bf(gelu_exact(a01[r]));
            Hsh[(16 + m) * HST + nb0 + col]      = f2bf(gelu_exact(a10[r]));
            Hsh[(16 + m) * HST + nb0 + 16 + col] = f2bf(gelu_exact(a11[r]));
        }
    }
    __syncthreads();

    // ------- phase 3: GEMM2 [32x256]x[256x64] + residual, all 8 waves ------
    // 2 M-tiles x 4 N-tiles = 8 wave tasks
    {
        const int mt = wv >> 2;                          // M-tile 0/1
        const int cb = (wv & 3) * 16;                    // output column block
        const float bv = b2[cb + col];
        v8f acc;
        #pragma unroll
        for (int i = 0; i < 8; ++i) acc[i] = bv;

        #pragma unroll
        for (int ks = 0; ks < HID / 32; ++ks) {          // 8 K-steps
            const int kO = ks * 32 + hi * 8;
            const unsigned short* ap = &Hsh[(mt * 16 + col) * HST + kO];
            v16bf A = pack_v16bf(*(const uint4*)ap, *(const uint4*)(ap + 16));
            const unsigned short* bp = W2t + (size_t)(cb + col) * HID + ks * 32 + hi * 8;
            v16bf Bv = pack_v16bf(*(const uint4*)bp, *(const uint4*)(bp + 16));
            acc = WMMA_BF16(A, Bv, acc);
        }

        #pragma unroll
        for (int r = 0; r < 8; ++r) {
            const int m = mt * 16 + r + hi * 8;
            const size_t gi = ((size_t)tile * TILE_M + m) * CH + cb + col;
            out[gi] = x[gi] + acc[r];                    // residual (f32 x reload, L2 hot)
        }
    }
}

// ------------------------------- launcher ---------------------------------
extern "C" void kernel_launch(void* const* d_in, const int* in_sizes, int n_in,
                              void* d_out, int out_size, void* d_ws, size_t ws_size,
                              hipStream_t stream) {
    const float* x      = (const float*)d_in[0];
    const float* conv_w = (const float*)d_in[1];
    const float* W1     = (const float*)d_in[2];
    const float* b1     = (const float*)d_in[3];
    const float* W2     = (const float*)d_in[4];
    const float* b2     = (const float*)d_in[5];

    unsigned short* W1t = (unsigned short*)d_ws;           // 81920 bf16
    unsigned short* W2t = W1t + KIN * HID;                 // 16384 bf16

    // 1) convert + transpose weights to bf16 [N][K] layout (cheap, every call)
    const int prep_elems = KIN * HID + HID * CH;           // 98304
    nca_prep_weights<<<(prep_elems + 255) / 256, 256, 0, stream>>>(W1, W2, W1t, W2t);

    // 2) fused NCA step: 8192 tiles of 32 pixels, 256 threads (8 wave32) each
    const int n_tiles = (16 * IMG_H * IMG_W) / TILE_M;     // 8192
    nca_step_kernel<<<n_tiles, 256, 0, stream>>>(x, conv_w, b1, b2, W1t, W2t,
                                                 (float*)d_out);
}
